// NvWaveNet_36971078484650
// MI455X (gfx1250) — matrix operand order repaired and checked by reading, not compiled, not run on hardware
//
#include <hip/hip_runtime.h>

#define LAYERS 20
#define LPS 10
#define B_   8
#define T_   16000
#define RES  64
#define GATE 128
#define SKIPC 256
#define OUTC 256

typedef __attribute__((ext_vector_type(16))) __bf16 v16bf;
typedef __attribute__((ext_vector_type(8)))  float  v8f;
typedef __attribute__((ext_vector_type(4)))  unsigned v4u;
typedef __attribute__((ext_vector_type(4)))  int      v4i;
typedef __attribute__((ext_vector_type(8)))  int      v8i;

#ifndef __has_builtin
#define __has_builtin(x) 0
#endif
#if __has_builtin(__builtin_amdgcn_tensor_load_to_lds) && \
    __has_builtin(__builtin_amdgcn_s_wait_tensorcnt)
#define HAVE_TDM 1
#else
#define HAVE_TDM 0
#endif

static __device__ __forceinline__ __bf16 f2bf(float f) {
  union { float f; unsigned u; } v; v.f = f;
  unsigned r = v.u + 0x7FFFu + ((v.u >> 16) & 1u);   // round-to-nearest-even
  unsigned short h = (unsigned short)(r >> 16);
  return __builtin_bit_cast(__bf16, h);
}

static __device__ __forceinline__ v8f wmma_bf16(v16bf a, v16bf b, v8f c) {
  return __builtin_amdgcn_wmma_f32_16x16x32_bf16(false, a, false, b, (short)0, c,
                                                 false, false);
}

#if HAVE_TDM
// Issue a TDM 2D tile load: `rows` rows of 128 bytes (64 bf16) from a
// contiguous [rows][64] bf16 region at `src` into LDS byte address `ldsb`.
// Descriptor per CDNA5 ISA ch.8: data_size=8B units -> 16 units per row.
static __device__ __forceinline__ void tdm_load_tile(const void* src,
                                                     unsigned ldsb,
                                                     unsigned rows) {
  const unsigned long long ga = (unsigned long long)src;
  const unsigned rowu = 16u;  // 128B row in 8-byte units
  v4u g0;
  g0.x = 1u;                                   // count=1, user mode
  g0.y = ldsb;                                 // lds_addr
  g0.z = (unsigned)(ga & 0xFFFFFFFFu);         // global_addr[31:0]
  g0.w = (unsigned)((ga >> 32) & 0x01FFFFFFu)  // global_addr[56:32]
         | 0x80000000u;                        // type=2 ("image")
  v8i g1;
  g1[0] = (int)(3u << 16);                     // data_size = 8B, no multicast
  g1[1] = (int)((rowu & 0xFFFFu) << 16);       // tensor_dim0 lo16
  g1[2] = (int)((rows & 0xFFFFu) << 16);       // tensor_dim0 hi16 | tensor_dim1 lo16
  g1[3] = (int)((rowu & 0xFFFFu) << 16);       // tensor_dim1 hi16 | tile_dim0
  g1[4] = (int)(rows & 0xFFFFu);               // tile_dim1 | tile_dim2=0
  g1[5] = (int)rowu;                           // tensor_dim0_stride lo32
  g1[6] = 0;                                   // stride hi16 | dim1_stride lo16
  g1[7] = 0;
  v4i z4 = {0, 0, 0, 0};
#if defined(__clang_major__) && (__clang_major__ >= 23)
  v8i z8 = {0, 0, 0, 0, 0, 0, 0, 0};
  __builtin_amdgcn_tensor_load_to_lds(g0, g1, z4, z4, z8, 0);
#else
  __builtin_amdgcn_tensor_load_to_lds(g0, g1, z4, z4, 0);
#endif
}
#endif

// ---------------------------------------------------------------------------
// Weight prep: f32 row-major (nmat, M, K) -> bf16 A-fragment layout.
// Fragment f = mtile*(K/32)+kstep; within fragment: lane 0..31, elem 0..15.
// 16-bit A 16x32 layout used here (contiguous k): m = lane%16, k = kstep*32 + (lane/16)*16 + e
// ---------------------------------------------------------------------------
__global__ void prep_kernel(const float* __restrict__ src, __bf16* __restrict__ dst,
                            int M, int K, int nmat) {
  int idx = blockIdx.x * 256 + threadIdx.x;
  int per = M * K;
  if (idx >= per * nmat) return;
  int mat  = idx / per;
  int r    = idx % per;
  int frag = r >> 9;            // /(32*16)
  int lane = (r >> 4) & 31;
  int e    = r & 15;
  int ksteps = K >> 5;
  int mtile = frag / ksteps;
  int kstep = frag % ksteps;
  int m = mtile * 16 + (lane & 15);
  int k = kstep * 32 + ((lane >> 4) << 4) + e;
  dst[idx] = f2bf(src[(size_t)mat * per + (size_t)m * K + k]);
}

// ---------------------------------------------------------------------------
// First causal conv: h = 2*(W_pre0 @ shift(x,1) + b_pre0), time-major outputs.
// ---------------------------------------------------------------------------
__global__ __launch_bounds__(256) void pre_kernel(
    const float* __restrict__ x,        // [B][256][T]
    const __bf16* __restrict__ wpre,    // frag 64x256
    const float* __restrict__ b_pre,    // [64]
    float* __restrict__ h32,            // [B][T][64]
    __bf16* __restrict__ hbf_out)       // [B][T][64]
{
  __shared__ __align__(32) __bf16 lds_x[64 * 256];
  const int tid = threadIdx.x;
  const int wave = tid >> 5, lane = tid & 31;
  const int nlo = lane & 15, hi = lane >> 4;
  const int b  = blockIdx.x / (T_ / 64);
  const int t0 = (blockIdx.x % (T_ / 64)) * 64;

  {
    const int t = tid & 63;
    const int ts = t0 + t - 1;
    for (int c = (tid >> 6); c < 256; c += 4) {
      float v = (ts >= 0) ? x[((size_t)(b * 256 + c)) * T_ + ts] : 0.0f;
      lds_x[t * 256 + c] = f2bf(v);
    }
  }
  __syncthreads();

  for (int s = 0; s < 2; ++s) {
    const int tt = wave * 2 + s;         // 16 tiles: 4 mtiles x 4 ntiles
    const int mt = tt >> 2, nt = tt & 3;
    v8f acc = {};
#pragma unroll
    for (int ks = 0; ks < 8; ++ks) {
      v16bf af = *(const v16bf*)(wpre + ((size_t)(mt * 8 + ks) * 32 + lane) * 16);
      v16bf bf = *(const v16bf*)(lds_x + (nt * 16 + nlo) * 256 + ks * 32 + hi * 16);
      acc = wmma_bf16(af, bf, acc);
    }
    const int t = t0 + nt * 16 + nlo;
    const size_t rowb = ((size_t)(b * T_ + t)) * 64;
#pragma unroll
    for (int r = 0; r < 8; ++r) {
      const int c = mt * 16 + hi * 8 + r;
      float hv = 2.0f * (acc[r] + b_pre[c]);
      h32[rowb + c] = hv;
      hbf_out[rowb + c] = f2bf(hv);
    }
  }
}

// ---------------------------------------------------------------------------
// One WaveNet layer, fully fused: gate matmul -> tanh*sigmoid -> res + skip.
// Activation tiles arrive in LDS via the Tensor Data Mover when available.
// ---------------------------------------------------------------------------
__global__ __launch_bounds__(256) void layer_kernel(
    const __bf16* __restrict__ hbf_in,   // [B][T][64]
    __bf16* __restrict__ hbf_out,        // [B][T][64]
    float* __restrict__ h32,             // [B][T][64] in-place residual
    float* __restrict__ skip32,          // [B][T][256] accumulated
    const __bf16* __restrict__ wdp,      // frag 128x64
    const __bf16* __restrict__ wdc,      // frag 128x64
    const float* __restrict__ bias_h,    // [128]
    const __bf16* __restrict__ wres,     // frag 64x64
    const float* __restrict__ b_res,     // [64]
    const __bf16* __restrict__ wskip,    // frag 256x64
    const float* __restrict__ b_skip,    // [256]
    int d)
{
  __shared__ __align__(32) __bf16 lds_hc[64 * 64];
  __shared__ __align__(32) __bf16 lds_hp[64 * 64];
  __shared__ __align__(32) float  lds_a[128 * 64];
  __shared__ __align__(32) __bf16 lds_g[64 * 64];

  const int tid = threadIdx.x;
  const int wave = tid >> 5, lane = tid & 31;
  const int nlo = lane & 15, hi = lane >> 4;
  const int b  = blockIdx.x / (T_ / 64);
  const int t0 = (blockIdx.x % (T_ / 64)) * 64;

#if HAVE_TDM
  // TDM tile fills: one wave issues the DMA, waits TENSORcnt, then the
  // workgroup barrier publishes the LDS tiles to all waves.
  if (wave == 0) {
    tdm_load_tile(hbf_in + ((size_t)(b * T_ + t0)) * 64,
                  (unsigned)(size_t)(&lds_hc[0]), 64);
    if (t0 >= d)
      tdm_load_tile(hbf_in + ((size_t)(b * T_ + t0 - d)) * 64,
                    (unsigned)(size_t)(&lds_hp[0]), 64);
    __builtin_amdgcn_s_wait_tensorcnt(0);
  }
  if (t0 < d) {  // causal edge: manual zero-padded rows for the shifted tile
    const int row = tid >> 2, q = tid & 3;
    uint4 z0 = {0, 0, 0, 0}, z1 = {0, 0, 0, 0};
    const int ts = t0 + row - d;
    if (ts >= 0) {
      const uint4* sp =
          (const uint4*)(hbf_in + ((size_t)(b * T_ + ts)) * 64 + q * 16);
      z0 = sp[0]; z1 = sp[1];
    }
    uint4* dp = (uint4*)(lds_hp) + (row * 8 + q * 2);
    dp[0] = z0; dp[1] = z1;
  }
#else
  // Cooperative time-major tile loads; causal shift = row offset w/ zero fill.
  {
    const int row = tid >> 2, q = tid & 3;     // 4 threads x 2x b128 per 128B row
    const uint4* sc =
        (const uint4*)(hbf_in + ((size_t)(b * T_ + t0 + row)) * 64 + q * 16);
    uint4* dc = (uint4*)(lds_hc) + (row * 8 + q * 2);
    dc[0] = sc[0]; dc[1] = sc[1];

    uint4 z0 = {0, 0, 0, 0}, z1 = {0, 0, 0, 0};
    const int ts = t0 + row - d;
    if (ts >= 0) {
      const uint4* sp =
          (const uint4*)(hbf_in + ((size_t)(b * T_ + ts)) * 64 + q * 16);
      z0 = sp[0]; z1 = sp[1];
    }
    uint4* dp = (uint4*)(lds_hp) + (row * 8 + q * 2);
    dp[0] = z0; dp[1] = z1;
  }
#endif
  if (t0 + 64 < T_)  // warm L2/WGP$ for next tile (global_prefetch_b8)
    __builtin_prefetch(hbf_in + ((size_t)(b * T_ + t0 + 64)) * 64 + tid * 16, 0, 0);
  __syncthreads();

  // Stage 1: a = Wdp @ h_pre + Wdc @ h_cur   (128 x 64 times)
  {
    const int mt = wave;                  // each wave owns one 16-row M tile
    for (int nt = 0; nt < 4; ++nt) {
      v8f acc = {};
#pragma unroll
      for (int ks = 0; ks < 2; ++ks) {
        v16bf ap = *(const v16bf*)(wdp + ((size_t)(mt * 2 + ks) * 32 + lane) * 16);
        v16bf bp = *(const v16bf*)(lds_hp + (nt * 16 + nlo) * 64 + ks * 32 + hi * 16);
        acc = wmma_bf16(ap, bp, acc);
        v16bf ac = *(const v16bf*)(wdc + ((size_t)(mt * 2 + ks) * 32 + lane) * 16);
        v16bf bc = *(const v16bf*)(lds_hc + (nt * 16 + nlo) * 64 + ks * 32 + hi * 16);
        acc = wmma_bf16(ac, bc, acc);
      }
      const int n = nt * 16 + nlo;
#pragma unroll
      for (int r = 0; r < 8; ++r)
        lds_a[(mt * 16 + hi * 8 + r) * 64 + n] = acc[r];
    }
  }
  __syncthreads();

  // Gate: g = tanh(a[:64]) * sigmoid(a[64:]), store time-major bf16
  for (int i = tid; i < 64 * 64; i += 256) {
    const int c = i >> 6, t = i & 63;
    const float at = lds_a[c * 64 + t] + bias_h[c];
    const float as = lds_a[(c + 64) * 64 + t] + bias_h[c + 64];
    const float gv = tanhf(at) * (1.0f / (1.0f + __expf(-as)));
    lds_g[t * 64 + c] = f2bf(gv);
  }
  __syncthreads();

  // Stage 2: res = Wres @ g ; h += res (f32), refresh bf16 copy
  for (int s = 0; s < 2; ++s) {
    const int tt = wave * 2 + s;          // 16 tiles: 4 mt x 4 nt
    const int mt = tt >> 2, nt = tt & 3;
    v8f acc = {};
#pragma unroll
    for (int ks = 0; ks < 2; ++ks) {
      v16bf af = *(const v16bf*)(wres + ((size_t)(mt * 2 + ks) * 32 + lane) * 16);
      v16bf bf = *(const v16bf*)(lds_g + (nt * 16 + nlo) * 64 + ks * 32 + hi * 16);
      acc = wmma_bf16(af, bf, acc);
    }
    const int t = t0 + nt * 16 + nlo;
    const size_t rowb = ((size_t)(b * T_ + t)) * 64;
#pragma unroll
    for (int r = 0; r < 8; ++r) {
      const int c = mt * 16 + hi * 8 + r;
      float hv = h32[rowb + c] + acc[r] + b_res[c];
      h32[rowb + c] = hv;
      hbf_out[rowb + c] = f2bf(hv);
    }
  }

  // Stage 3: skip += Wskip @ g
  for (int s = 0; s < 8; ++s) {
    const int tt = wave * 8 + s;          // 64 tiles: 16 mt x 4 nt
    const int mt = tt >> 2, nt = tt & 3;
    v8f acc = {};
#pragma unroll
    for (int ks = 0; ks < 2; ++ks) {
      v16bf af = *(const v16bf*)(wskip + ((size_t)(mt * 2 + ks) * 32 + lane) * 16);
      v16bf bf = *(const v16bf*)(lds_g + (nt * 16 + nlo) * 64 + ks * 32 + hi * 16);
      acc = wmma_bf16(af, bf, acc);
    }
    const int t = t0 + nt * 16 + nlo;
    const size_t sb = ((size_t)(b * T_ + t)) * 256;
#pragma unroll
    for (int r = 0; r < 8; ++r) {
      const int c = mt * 16 + hi * 8 + r;
      skip32[sb + c] += acc[r] + b_skip[c];
    }
  }
}

// ---------------------------------------------------------------------------
// Head: relu -> W_relu -> relu -> W_out -> softmax over channels.
// ---------------------------------------------------------------------------
__global__ __launch_bounds__(256) void head_kernel(
    const float* __restrict__ skip32,   // [B][T][256]
    const __bf16* __restrict__ wrelu,   // frag 256x256
    const float* __restrict__ b_relu,
    const __bf16* __restrict__ wout,    // frag 256x256
    const float* __restrict__ b_out,
    float* __restrict__ out)            // [B][256][T]
{
  __shared__ __align__(32) __bf16 bufA[32 * 256];
  __shared__ __align__(32) __bf16 bufB[32 * 256];
  __shared__ __align__(32) float  bufC[32 * 256];

  const int tid = threadIdx.x;
  const int wave = tid >> 5, lane = tid & 31;
  const int nlo = lane & 15, hi = lane >> 4;
  const int b  = blockIdx.x / (T_ / 32);
  const int t0 = (blockIdx.x % (T_ / 32)) * 32;

  for (int i = tid; i < 32 * 256; i += 256) {
    float v = skip32[((size_t)(b * T_ + t0)) * 256 + i];
    bufA[i] = f2bf(v > 0.0f ? v : 0.0f);
  }
  __syncthreads();

  // o1 = relu(W_relu @ relu(skip) + b_relu)
  for (int s = 0; s < 4; ++s) {
    const int tt = wave * 4 + s;          // 32 tiles: 16 mt x 2 nt
    const int mt = tt >> 1, nt = tt & 1;
    v8f acc = {};
#pragma unroll
    for (int ks = 0; ks < 8; ++ks) {
      v16bf af = *(const v16bf*)(wrelu + ((size_t)(mt * 8 + ks) * 32 + lane) * 16);
      v16bf bf = *(const v16bf*)(bufA + (nt * 16 + nlo) * 256 + ks * 32 + hi * 16);
      acc = wmma_bf16(af, bf, acc);
    }
    const int t = nt * 16 + nlo;
#pragma unroll
    for (int r = 0; r < 8; ++r) {
      const int c = mt * 16 + hi * 8 + r;
      float o = acc[r] + b_relu[c];
      bufB[t * 256 + c] = f2bf(o > 0.0f ? o : 0.0f);
    }
  }
  __syncthreads();

  // logits = W_out @ o1 + b_out
  for (int s = 0; s < 4; ++s) {
    const int tt = wave * 4 + s;
    const int mt = tt >> 1, nt = tt & 1;
    v8f acc = {};
#pragma unroll
    for (int ks = 0; ks < 8; ++ks) {
      v16bf af = *(const v16bf*)(wout + ((size_t)(mt * 8 + ks) * 32 + lane) * 16);
      v16bf bf = *(const v16bf*)(bufB + (nt * 16 + nlo) * 256 + ks * 32 + hi * 16);
      acc = wmma_bf16(af, bf, acc);
    }
    const int t = nt * 16 + nlo;
#pragma unroll
    for (int r = 0; r < 8; ++r) {
      const int c = mt * 16 + hi * 8 + r;
      bufC[t * 256 + c] = acc[r] + b_out[c];
    }
  }
  __syncthreads();

  if (tid < 32) {
    const float* row = bufC + tid * 256;
    float m = row[0];
    for (int c = 1; c < 256; ++c) m = fmaxf(m, row[c]);
    float sum = 0.0f;
    for (int c = 0; c < 256; ++c) sum += __expf(row[c] - m);
    const float inv = 1.0f / sum;
    for (int c = 0; c < 256; ++c)
      out[((size_t)(b * 256 + c)) * T_ + (t0 + tid)] = __expf(row[c] - m) * inv;
  }
}

// ---------------------------------------------------------------------------
extern "C" void kernel_launch(void* const* d_in, const int* in_sizes, int n_in,
                              void* d_out, int out_size, void* d_ws, size_t ws_size,
                              hipStream_t stream) {
  const float* x      = (const float*)d_in[0];
  const float* w_pre0 = (const float*)d_in[1];
  const float* b_pre0 = (const float*)d_in[2];
  const float* w_dpre = (const float*)d_in[3];
  const float* w_dcur = (const float*)d_in[4];
  const float* bias_h = (const float*)d_in[5];
  const float* w_res  = (const float*)d_in[6];
  const float* b_res  = (const float*)d_in[7];
  const float* w_skip = (const float*)d_in[8];
  const float* b_skip = (const float*)d_in[9];
  const float* w_relu = (const float*)d_in[10];
  const float* b_relu = (const float*)d_in[11];
  const float* w_out  = (const float*)d_in[12];
  const float* b_out  = (const float*)d_in[13];
  float* out = (float*)d_out;

  char* ws = (char*)d_ws;
  const size_t HB = (size_t)B_ * T_ * RES;     // h elements
  const size_t SB = (size_t)B_ * T_ * SKIPC;   // skip elements
  float*  h32    = (float*)ws;                 // HB * 4
  __bf16* hbfA   = (__bf16*)(ws + HB * 4);     // HB * 2
  __bf16* hbfB   = (__bf16*)(ws + HB * 6);     // HB * 2
  float*  skip32 = (float*)(ws + HB * 8);      // SB * 4
  char* wp = ws + HB * 8 + SB * 4;
  __bf16* f_pre0 = (__bf16*)wp; wp += (size_t)64 * 256 * 2;
  __bf16* f_dpre = (__bf16*)wp; wp += (size_t)LAYERS * 128 * 64 * 2;
  __bf16* f_dcur = (__bf16*)wp; wp += (size_t)LAYERS * 128 * 64 * 2;
  __bf16* f_res  = (__bf16*)wp; wp += (size_t)LAYERS * 64 * 64 * 2;
  __bf16* f_skip = (__bf16*)wp; wp += (size_t)LAYERS * 256 * 64 * 2;
  __bf16* f_relu = (__bf16*)wp; wp += (size_t)256 * 256 * 2;
  __bf16* f_out  = (__bf16*)wp; wp += (size_t)256 * 256 * 2;

  hipMemsetAsync(skip32, 0, SB * 4, stream);

  auto gprep = [&](const float* s, __bf16* dst, int M, int K, int nm) {
    int total = M * K * nm;
    prep_kernel<<<(total + 255) / 256, 256, 0, stream>>>(s, dst, M, K, nm);
  };
  gprep(w_pre0, f_pre0, 64, 256, 1);
  gprep(w_dpre, f_dpre, 128, 64, LAYERS);
  gprep(w_dcur, f_dcur, 128, 64, LAYERS);
  gprep(w_res,  f_res,  64, 64, LAYERS);
  gprep(w_skip, f_skip, 256, 64, LAYERS);
  gprep(w_relu, f_relu, 256, 256, 1);
  gprep(w_out,  f_out,  256, 256, 1);

  dim3 blk(256);
  pre_kernel<<<B_ * (T_ / 64), blk, 0, stream>>>(x, f_pre0, b_pre0, h32, hbfA);

  for (int l = 0; l < LAYERS; ++l) {
    const int d = 1 << (l % LPS);
    const __bf16* hin = (l & 1) ? hbfB : hbfA;
    __bf16* hout      = (l & 1) ? hbfA : hbfB;
    layer_kernel<<<B_ * (T_ / 64), blk, 0, stream>>>(
        hin, hout, h32, skip32,
        f_dpre + (size_t)l * 128 * 64, f_dcur + (size_t)l * 128 * 64,
        bias_h + (size_t)l * 128,
        f_res + (size_t)l * 64 * 64, b_res + (size_t)l * 64,
        f_skip + (size_t)l * 256 * 64, b_skip + (size_t)l * 256,
        d);
  }

  head_kernel<<<B_ * (T_ / 32), blk, 0, stream>>>(skip32, f_relu, b_relu,
                                                  f_out, b_out, out);
}
